// custom_gnn_3831110828322
// MI455X (gfx1250) — compile-verified, matrix operand order checked by reading, not run on hardware
//
#include <hip/hip_runtime.h>
#include <math.h>

#define NN 32768
#define EE 262144
#define LL 4
#define HH 8
#define DD 64
#define HIDD 512
#define INDIM 386
#define INPAD 416          // 386 padded to multiple of 32
#define OUTDIM 384
#define EDIMM 32
#define MAXSP 4

typedef __attribute__((ext_vector_type(16))) __bf16 v16bf;
typedef __attribute__((ext_vector_type(8)))  float  v8f;

union Frag { v16bf v; unsigned int u[8]; };

static __device__ inline unsigned short f2bf(float f) {
  unsigned int u = __float_as_uint(f);
  unsigned int r = u + 0x7fffu + ((u >> 16) & 1u);   // round-to-nearest-even
  return (unsigned short)(r >> 16);
}

// monotonic float<->int mapping for atomic max on signed ints
static __host__ __device__ inline int f32_ord(float f) {
  int b; __builtin_memcpy(&b, &f, 4);
  return b >= 0 ? b : (int)(0x80000000u - (unsigned)b);
}
static __device__ inline float ord_f32(int o) {
  int b = (o >= 0) ? o : (int)(0x80000000u - (unsigned)o);
  float f; __builtin_memcpy(&f, &b, 4);
  return f;
}

// CDNA5 async global->LDS copy (ASYNCcnt-tracked, bypasses VGPRs).
// VDST = LDS byte address (low 32 bits of flat shared address), VADDR = 64-bit global addr.
#define ASYNC_B128(ldsaddr, gptr)                                              \
  asm volatile("global_load_async_to_lds_b128 %0, %1, off"                     \
               :: "v"(ldsaddr), "v"((unsigned long long)(size_t)(gptr))        \
               : "memory")
#define WAIT_ASYNC(n) asm volatile("s_wait_asynccnt " #n ::: "memory")

#define LDSS 40   // padded LDS stride (bf16 elems): 80B keeps 16B alignment, 16 rows -> 16 banks

// Assemble a 16x32 bf16 fragment per ISA 7.12.2 (16-bit A/B layout):
// lane<16: row=lane,     K={0..7} in v0..3 (2/vgpr), {16..23} in v4..7
// lane>=16: row=lane-16, K={8..15} in v0..3,          {24..31} in v4..7
static __device__ inline v16bf load_frag(const unsigned short* sm, int rowbase, int lane) {
  Frag f;
  const int kb = (lane & 16) ? 8 : 0;
#pragma unroll
  for (int j = 0; j < 4; ++j)
    __builtin_memcpy(&f.u[j], sm + rowbase + kb + 2 * j, 4);
#pragma unroll
  for (int j = 0; j < 4; ++j)
    __builtin_memcpy(&f.u[4 + j], sm + rowbase + 16 + kb + 2 * j, 4);
  return f.v;
}

// C[M,N] = A[M,Kpad] @ Bt[N,Kpad]^T + bias ; bf16 in (pre-converted), f32 out
// (+ optional fused bf16 mirror of C for the next GEMM's A operand).
// 256 threads = 8 waves; block tile 128x128; wave tile 32x64 (2x4 WMMA tiles).
// Double-buffered LDS staged with async B128 copies overlapping the WMMAs.
__global__ __launch_bounds__(256) void gemm_bf16_wmma(
    const unsigned short* __restrict__ A,   // [M][Kpad] bf16 row-major
    const unsigned short* __restrict__ Bt,  // [N][Kpad] bf16 (B pre-transposed)
    const float* __restrict__ bias, float* __restrict__ C,
    unsigned short* __restrict__ Cbf,       // optional [M][N] bf16 mirror (may be null)
    int Kpad, int N)
{
  __shared__ unsigned short As[2][128 * LDSS];
  __shared__ unsigned short Bs[2][128 * LDSS];
  const int tid  = threadIdx.x;
  const int lane = tid & 31;
  const int wave = tid >> 5;
  const int wm = wave & 3;   // 4 waves along M (32 rows each)
  const int wn = wave >> 2;  // 2 waves along N (64 cols each)
  const size_t mBase = (size_t)blockIdx.y * 128;
  const size_t nBase = (size_t)blockIdx.x * 128;

  v8f acc[2][4];
#pragma unroll
  for (int a = 0; a < 2; ++a)
#pragma unroll
    for (int b = 0; b < 4; ++b)
#pragma unroll
      for (int i = 0; i < 8; ++i) acc[a][b][i] = 0.f;

  const int nkt = Kpad >> 5;

  // stage one 128x32 bf16 tile of A and of B: 512 x 16B chunks each, 2+2 per thread
  auto stage = [&](int buf, int k0) {
#pragma unroll
    for (int i = 0; i < 2; ++i) {
      int c = i * 256 + tid;          // 0..511
      int r = c >> 2, part = c & 3;   // 4 x 16B chunks per 64B row
      const unsigned short* ga = A  + (mBase + r) * (size_t)Kpad + k0 + part * 8;
      const unsigned short* gb = Bt + (nBase + r) * (size_t)Kpad + k0 + part * 8;
      unsigned la = (unsigned)(size_t)(const void*)&As[buf][r * LDSS + part * 8];
      unsigned lb = (unsigned)(size_t)(const void*)&Bs[buf][r * LDSS + part * 8];
      ASYNC_B128(la, ga);
      ASYNC_B128(lb, gb);
    }
  };

  auto compute = [&](int buf) {
    v16bf af[2], bfr[4];
#pragma unroll
    for (int tm = 0; tm < 2; ++tm)
      af[tm] = load_frag(&As[buf][0], (wm * 32 + tm * 16 + (lane & 15)) * LDSS, lane);
#pragma unroll
    for (int tn = 0; tn < 4; ++tn)
      bfr[tn] = load_frag(&Bs[buf][0], (wn * 64 + tn * 16 + (lane & 15)) * LDSS, lane);
#pragma unroll
    for (int tm = 0; tm < 2; ++tm)
#pragma unroll
      for (int tn = 0; tn < 4; ++tn)
        acc[tm][tn] = __builtin_amdgcn_wmma_f32_16x16x32_bf16(
            false, af[tm], false, bfr[tn], (short)0, acc[tm][tn], false, false);
  };

  stage(0, 0);
  int kt = 0;
  for (; kt < nkt - 1; ++kt) {            // steady state: prefetch next, compute current
    stage((kt & 1) ^ 1, (kt + 1) * 32);
    WAIT_ASYNC(4);                        // allow the 4 just-issued copies to stay in flight
    __syncthreads();
    compute(kt & 1);
    __syncthreads();                      // protect the buffer next stage() overwrites
  }
  WAIT_ASYNC(0);                          // last tile
  __syncthreads();
  compute(kt & 1);

  // epilogue per ISA C/D layout: vgpr i -> M=i (lanes 0-15) / M=i+8 (lanes 16-31), N=lane&15
#pragma unroll
  for (int tm = 0; tm < 2; ++tm) {
#pragma unroll
    for (int tn = 0; tn < 4; ++tn) {
      size_t row = mBase + wm * 32 + tm * 16 + ((lane & 16) ? 8 : 0);
      int    col = (int)nBase + wn * 64 + tn * 16 + (lane & 15);
      float  bv  = bias[col];
#pragma unroll
      for (int i = 0; i < 8; ++i) {
        float val = acc[tm][tn][i] + bv;
        C[(row + i) * N + col] = val;
        if (Cbf) Cbf[(row + i) * N + col] = f2bf(val);
      }
    }
  }
}

// dst[m][kp] = bf16(src[m][k]) with zero K-padding (row-major activations)
__global__ void cvt_rows_kernel(const float* __restrict__ src, unsigned short* __restrict__ dst,
                                int K, int Kpad, size_t total) {
  size_t t = (size_t)blockIdx.x * 256 + threadIdx.x;
  if (t >= total) return;
  size_t m = t / (size_t)Kpad;
  int kp   = (int)(t - m * Kpad);
  dst[t] = (kp < K) ? f2bf(src[m * K + kp]) : (unsigned short)0;
}

// Wt[n][kp] = bf16(W[k][n]) with zero K-padding (weights transposed once)
__global__ void cvt_transpose_kernel(const float* __restrict__ W, unsigned short* __restrict__ Wt,
                                     int K, int N, int Kpad) {
  size_t t = (size_t)blockIdx.x * 256 + threadIdx.x;
  if (t >= (size_t)N * Kpad) return;
  size_t n = t / (size_t)Kpad;
  int kp   = (int)(t - n * Kpad);
  Wt[t] = (kp < K) ? f2bf(W[(size_t)kp * N + n]) : (unsigned short)0;
}

// eproj[sp][j] = sum_c Etab[sp][c] * We[c][j] + be[j]   (4 x 32 x 512, tiny)
__global__ void eproj_kernel(const float* __restrict__ Etab, const float* __restrict__ We,
                             const float* __restrict__ be, float* __restrict__ eproj)
{
  int t = blockIdx.x * 256 + threadIdx.x;
  if (t >= MAXSP * HIDD) return;
  int sp = t >> 9, j = t & (HIDD - 1);
  float acc = be[j];
#pragma unroll
  for (int c = 0; c < EDIMM; ++c) acc += Etab[sp * EDIMM + c] * We[(size_t)c * HIDD + j];
  eproj[t] = acc;
}

__global__ void fill_f32_kernel(float* p, float v, size_t n) {
  size_t i = (size_t)blockIdx.x * 256 + threadIdx.x;
  if (i < n) p[i] = v;
}
__global__ void fill_i32_kernel(int* p, int v, size_t n) {
  size_t i = (size_t)blockIdx.x * 256 + threadIdx.x;
  if (i < n) p[i] = v;
}

// alpha[e,h] = scale * dot(q[dst,h,:], k[src,h,:]+eproj[id,h,:]); atomic max into amax[dst,h]
__global__ __launch_bounds__(256) void edge_alpha_kernel(
    const float* __restrict__ q, const float* __restrict__ k, const float* __restrict__ eproj,
    const int* __restrict__ ei, const int* __restrict__ ids,
    float* __restrict__ alpha, int* __restrict__ amax)
{
  int t = blockIdx.x * 256 + threadIdx.x;
  if (t >= EE * HH) return;
  int e = t >> 3, hh = t & 7;
  int src = ei[e], dst = ei[EE + e], id = ids[e];
  const float4* qp = (const float4*)(q + (size_t)dst * HIDD + hh * DD);
  const float4* kp = (const float4*)(k + (size_t)src * HIDD + hh * DD);
  const float4* ep = (const float4*)(eproj + (size_t)id * HIDD + hh * DD);
  float acc = 0.f;
#pragma unroll
  for (int i = 0; i < 16; ++i) {
    float4 a = qp[i], b = kp[i], c = ep[i];
    acc += a.x * (b.x + c.x) + a.y * (b.y + c.y) + a.z * (b.z + c.z) + a.w * (b.w + c.w);
  }
  acc *= 0.125f;  // 1/sqrt(64)
  alpha[t] = acc;
  atomicMax(&amax[dst * HH + hh], f32_ord(acc));
}

// ex = exp(alpha - amax[dst]); atomic add into denom[dst,h]; alpha <- ex
__global__ __launch_bounds__(256) void edge_norm_kernel(
    const int* __restrict__ ei, const int* __restrict__ amax,
    float* __restrict__ alpha, float* __restrict__ denom)
{
  int t = blockIdx.x * 256 + threadIdx.x;
  if (t >= EE * HH) return;
  int e = t >> 3, hh = t & 7;
  int dst = ei[EE + e];
  float m = ord_f32(amax[dst * HH + hh]);
  float ex = __expf(alpha[t] - m);
  alpha[t] = ex;
  atomicAdd(&denom[dst * HH + hh], ex);
}

// out[dst,h,:] += (ex/denom[dst,h]) * (v[src,h,:] + eproj[id,h,:])
__global__ __launch_bounds__(256) void edge_scatter_kernel(
    const float* __restrict__ v, const float* __restrict__ eproj,
    const int* __restrict__ ei, const int* __restrict__ ids,
    const float* __restrict__ alpha, const float* __restrict__ denom,
    float* __restrict__ out)
{
  int t = blockIdx.x * 256 + threadIdx.x;
  if (t >= EE * HH) return;
  int e = t >> 3, hh = t & 7;
  int src = ei[e], dst = ei[EE + e], id = ids[e];
  float a = alpha[t] / denom[dst * HH + hh];
  const float4* vp = (const float4*)(v + (size_t)src * HIDD + hh * DD);
  const float4* ep = (const float4*)(eproj + (size_t)id * HIDD + hh * DD);
  float* op = out + (size_t)dst * HIDD + hh * DD;
#pragma unroll
  for (int i = 0; i < 16; ++i) {
    float4 b = vp[i], c = ep[i];
    atomicAdd(op + 4 * i + 0, a * (b.x + c.x));
    atomicAdd(op + 4 * i + 1, a * (b.y + c.y));
    atomicAdd(op + 4 * i + 2, a * (b.z + c.z));
    atomicAdd(op + 4 * i + 3, a * (b.w + c.w));
  }
}

// beta gate + residual + LayerNorm, one node per 256-thread block (2 elems/thread).
// Writes both f32 h and its bf16 mirror (fused, saves a full conversion pass).
__global__ __launch_bounds__(256) void node_update_kernel(
    float* __restrict__ h, unsigned short* __restrict__ hbf,
    const float* __restrict__ attn, const float* __restrict__ xr,
    const float* __restrict__ Wb, const float* __restrict__ bb,
    const float* __restrict__ g, const float* __restrict__ b)
{
  __shared__ float red[256];
  const int n = blockIdx.x;
  const int t = threadIdx.x;
  const size_t base = (size_t)n * HIDD;
  float o0 = attn[base + t],       o1 = attn[base + 256 + t];
  float x0 = xr[base + t],         x1 = xr[base + 256 + t];
  float p = o0 * Wb[t] + x0 * Wb[HIDD + t] + (o0 - x0) * Wb[2 * HIDD + t]
          + o1 * Wb[256 + t] + x1 * Wb[HIDD + 256 + t] + (o1 - x1) * Wb[2 * HIDD + 256 + t];
  red[t] = p; __syncthreads();
  for (int st = 128; st > 0; st >>= 1) { if (t < st) red[t] += red[t + st]; __syncthreads(); }
  float beta = 1.f / (1.f + __expf(-(red[0] + bb[0])));
  __syncthreads();

  float y0 = h[base + t]       + beta * x0 + (1.f - beta) * o0;
  float y1 = h[base + 256 + t] + beta * x1 + (1.f - beta) * o1;
  red[t] = y0 + y1; __syncthreads();
  for (int st = 128; st > 0; st >>= 1) { if (t < st) red[t] += red[t + st]; __syncthreads(); }
  float mu = red[0] * (1.f / HIDD);
  __syncthreads();

  float d0 = y0 - mu, d1 = y1 - mu;
  red[t] = d0 * d0 + d1 * d1; __syncthreads();
  for (int st = 128; st > 0; st >>= 1) { if (t < st) red[t] += red[t + st]; __syncthreads(); }
  float rs = rsqrtf(red[0] * (1.f / HIDD) + 1e-5f);
  float r0 = d0 * rs * g[t]       + b[t];
  float r1 = d1 * rs * g[256 + t] + b[256 + t];
  h[base + t]         = r0;
  h[base + 256 + t]   = r1;
  hbf[base + t]       = f2bf(r0);
  hbf[base + 256 + t] = f2bf(r1);
}

extern "C" void kernel_launch(void* const* d_in, const int* in_sizes, int n_in,
                              void* d_out, int out_size, void* d_ws, size_t ws_size,
                              hipStream_t stream) {
  const float* emb   = (const float*)d_in[0];
  const int*   ei    = (const int*)  d_in[1];
  const int*   ids   = (const int*)  d_in[2];
  const float* Win   = (const float*)d_in[3];
  const float* bin_  = (const float*)d_in[4];
  const float* Etab  = (const float*)d_in[5];
  const float* Wq    = (const float*)d_in[6];
  const float* bq    = (const float*)d_in[7];
  const float* Wk    = (const float*)d_in[8];
  const float* bk    = (const float*)d_in[9];
  const float* Wv    = (const float*)d_in[10];
  const float* bv    = (const float*)d_in[11];
  const float* We    = (const float*)d_in[12];
  const float* be    = (const float*)d_in[13];
  const float* Ws    = (const float*)d_in[14];
  const float* bs    = (const float*)d_in[15];
  const float* Wbeta = (const float*)d_in[16];
  const float* bbeta = (const float*)d_in[17];
  const float* lng   = (const float*)d_in[18];
  const float* lnb   = (const float*)d_in[19];
  const float* Wout  = (const float*)d_in[20];
  const float* bout  = (const float*)d_in[21];

  // ---- workspace layout (f32 region, then bf16 region) ----
  float* ws   = (float*)d_ws;
  float* h    = ws;
  float* q    = h    + (size_t)NN * HIDD;
  float* k    = q    + (size_t)NN * HIDD;
  float* v    = k    + (size_t)NN * HIDD;
  float* sb   = v    + (size_t)NN * HIDD;
  float* attn = sb   + (size_t)NN * HIDD;
  float* alph = attn + (size_t)NN * HIDD;
  float* den  = alph + (size_t)EE * HH;
  int*   amax = (int*)(den + (size_t)NN * HH);
  float* epj  = (float*)(amax + (size_t)NN * HH);

  unsigned short* emb_bf = (unsigned short*)(epj + MAXSP * HIDD);
  unsigned short* h_bf   = emb_bf + (size_t)NN * INPAD;
  unsigned short* win_t  = h_bf   + (size_t)NN * HIDD;
  unsigned short* wqt    = win_t  + (size_t)HIDD * INPAD;           // 4 layers each below
  unsigned short* wkt    = wqt    + (size_t)LL * HIDD * HIDD;
  unsigned short* wvt    = wkt    + (size_t)LL * HIDD * HIDD;
  unsigned short* wst    = wvt    + (size_t)LL * HIDD * HIDD;
  unsigned short* wot    = wst    + (size_t)LL * HIDD * HIDD;       // [OUTDIM][HIDD]

  const dim3 blk(256);
  const int  egrid = (EE * HH + 255) / 256;
  const int  ngrid = (NN * HH + 255) / 256;
  const int  fgrid = (int)(((size_t)NN * HIDD + 255) / 256);
  const int  neg_ord = f32_ord(-INFINITY);

  // ---- one-time conversions: embeddings row-major bf16, weights transposed bf16 ----
  {
    size_t tot = (size_t)NN * INPAD;
    cvt_rows_kernel<<<(unsigned)((tot + 255) / 256), blk, 0, stream>>>(emb, emb_bf, INDIM, INPAD, tot);
    cvt_transpose_kernel<<<(HIDD * INPAD + 255) / 256, blk, 0, stream>>>(Win, win_t, INDIM, HIDD, INPAD);
    for (int l = 0; l < LL; ++l) {
      size_t o = (size_t)l * HIDD * HIDD;
      cvt_transpose_kernel<<<(HIDD * HIDD + 255) / 256, blk, 0, stream>>>(Wq + o, wqt + o, HIDD, HIDD, HIDD);
      cvt_transpose_kernel<<<(HIDD * HIDD + 255) / 256, blk, 0, stream>>>(Wk + o, wkt + o, HIDD, HIDD, HIDD);
      cvt_transpose_kernel<<<(HIDD * HIDD + 255) / 256, blk, 0, stream>>>(Wv + o, wvt + o, HIDD, HIDD, HIDD);
      cvt_transpose_kernel<<<(HIDD * HIDD + 255) / 256, blk, 0, stream>>>(Ws + o, wst + o, HIDD, HIDD, HIDD);
    }
    cvt_transpose_kernel<<<(OUTDIM * HIDD + 255) / 256, blk, 0, stream>>>(Wout, wot, HIDD, OUTDIM, HIDD);
  }

  // h = emb @ Win + bin   (fused bf16 mirror -> h_bf)
  gemm_bf16_wmma<<<dim3(HIDD / 128, NN / 128), blk, 0, stream>>>(
      emb_bf, win_t, bin_, h, h_bf, INPAD, HIDD);

  for (int l = 0; l < LL; ++l) {
    size_t o = (size_t)l * HIDD * HIDD;
    eproj_kernel<<<(MAXSP * HIDD + 255) / 256, blk, 0, stream>>>(
        Etab, We + (size_t)l * EDIMM * HIDD, be + (size_t)l * HIDD, epj);

    gemm_bf16_wmma<<<dim3(HIDD / 128, NN / 128), blk, 0, stream>>>(
        h_bf, wqt + o, bq + (size_t)l * HIDD, q, (unsigned short*)nullptr, HIDD, HIDD);
    gemm_bf16_wmma<<<dim3(HIDD / 128, NN / 128), blk, 0, stream>>>(
        h_bf, wkt + o, bk + (size_t)l * HIDD, k, (unsigned short*)nullptr, HIDD, HIDD);
    gemm_bf16_wmma<<<dim3(HIDD / 128, NN / 128), blk, 0, stream>>>(
        h_bf, wvt + o, bv + (size_t)l * HIDD, v, (unsigned short*)nullptr, HIDD, HIDD);
    gemm_bf16_wmma<<<dim3(HIDD / 128, NN / 128), blk, 0, stream>>>(
        h_bf, wst + o, bs + (size_t)l * HIDD, sb, (unsigned short*)nullptr, HIDD, HIDD);

    fill_i32_kernel<<<ngrid, blk, 0, stream>>>(amax, neg_ord, (size_t)NN * HH);
    fill_f32_kernel<<<ngrid, blk, 0, stream>>>(den, 0.f, (size_t)NN * HH);
    fill_f32_kernel<<<fgrid, blk, 0, stream>>>(attn, 0.f, (size_t)NN * HIDD);

    edge_alpha_kernel<<<egrid, blk, 0, stream>>>(q, k, epj, ei, ids, alph, amax);
    edge_norm_kernel<<<egrid, blk, 0, stream>>>(ei, amax, alph, den);
    edge_scatter_kernel<<<egrid, blk, 0, stream>>>(v, epj, ei, ids, alph, den, attn);

    node_update_kernel<<<NN, blk, 0, stream>>>(
        h, h_bf, attn, sb, Wbeta + (size_t)l * 3 * HIDD, bbeta + l,
        lng + (size_t)l * HIDD, lnb + (size_t)l * HIDD);
  }

  // out = h @ Wout + bout   (h_bf already produced by the last node_update)
  gemm_bf16_wmma<<<dim3(OUTDIM / 128, NN / 128), blk, 0, stream>>>(
      h_bf, wot, bout, (float*)d_out, (unsigned short*)nullptr, HIDD, OUTDIM);
}
